// MoEDecoderLayer_57758720196697
// MI455X (gfx1250) — matrix operand order repaired and checked by reading, not compiled, not run on hardware
//
#include <hip/hip_runtime.h>
#include <hip/hip_bf16.h>

typedef __attribute__((ext_vector_type(16))) __bf16 v16bf;
typedef __attribute__((ext_vector_type(2)))  __bf16 v2bf;
typedef __attribute__((ext_vector_type(8)))  float  v8f;

namespace {
constexpr int   cH = 2048;   // hidden
constexpr int   cI = 1024;   // intermediate
constexpr int   cE = 16;     // experts
constexpr int   cK = 2;      // top-k
constexpr int   cT = 4096;   // tokens (B*S)
constexpr float cEPS = 1e-6f;
constexpr int   MAXTILES = 160;  // sum ceil(cnt_e/64) <= 128+16 = 144
constexpr int   BM = 64, BN = 64, BK = 32, LDSP = 34; // LDS row stride (pad 2)
}

__device__ __forceinline__ unsigned short f2bf(float f) {
  unsigned int u = __float_as_uint(f);
  u += 0x7FFFu + ((u >> 16) & 1u);       // round-to-nearest-even
  return (unsigned short)(u >> 16);
}

// packed fp32x2 -> bf16x2 (single v_cvt_pk_bf16_f32 when available)
__device__ __forceinline__ unsigned int pkbf(float lo, float hi) {
#if __has_builtin(__builtin_amdgcn_cvt_pk_bf16_f32)
  v2bf r = __builtin_amdgcn_cvt_pk_bf16_f32(lo, hi);
  return __builtin_bit_cast(unsigned int, r);
#else
  return (unsigned)f2bf(lo) | ((unsigned)f2bf(hi) << 16);
#endif
}

union FragU { v16bf v; unsigned int u[8]; };

// ---------------------------------------------------------------- utilities
__global__ void init_kernel(int* counts) {
  int i = threadIdx.x;
  if (i < cE) counts[i] = 0;
}

__global__ void zero_kernel(float* p, size_t n4) {
  size_t i = (size_t)blockIdx.x * blockDim.x + threadIdx.x;
  if (i < n4) ((float4*)p)[i] = make_float4(0.f, 0.f, 0.f, 0.f);
}

__global__ void f32_to_bf16_kernel(const float* __restrict__ src,
                                   unsigned int* __restrict__ dst, size_t n2) {
  size_t i = (size_t)blockIdx.x * blockDim.x + threadIdx.x;
  if (i < n2) {
    float2 v = ((const float2*)src)[i];
    dst[i] = pkbf(v.x, v.y);
  }
}

// -------------------------------------------- fused RMSNorm + router (fp32)
__global__ void norm_gate_kernel(const float* __restrict__ x,
                                 const float* __restrict__ nw,
                                 const float* __restrict__ gw,
                                 unsigned short* __restrict__ xn,
                                 float* __restrict__ topkw,
                                 int* __restrict__ topkids,
                                 float* __restrict__ out_topkw,
                                 int* __restrict__ out_topkids,
                                 int* __restrict__ counts) {
  int t = blockIdx.x;
  int tid = threadIdx.x;
  const float* xr = x + (size_t)t * cH;

  __shared__ float red[256];
  float4 xa = *(const float4*)(xr + tid * 8);
  float4 xb = *(const float4*)(xr + tid * 8 + 4);
  float ss = xa.x * xa.x + xa.y * xa.y + xa.z * xa.z + xa.w * xa.w +
             xb.x * xb.x + xb.y * xb.y + xb.z * xb.z + xb.w * xb.w;
  red[tid] = ss;
  __syncthreads();
  for (int s = 128; s > 0; s >>= 1) {
    if (tid < s) red[tid] += red[tid + s];
    __syncthreads();
  }
  float rs = rsqrtf(red[0] * (1.0f / cH) + cEPS);

  // normalized activations in bf16 for the WMMA GEMMs (packed stores)
  {
    float4 wa = *(const float4*)(nw + tid * 8);
    float4 wb = *(const float4*)(nw + tid * 8 + 4);
    unsigned int* dst = (unsigned int*)(xn + (size_t)t * cH + tid * 8);
    dst[0] = pkbf(xa.x * rs * wa.x, xa.y * rs * wa.y);
    dst[1] = pkbf(xa.z * rs * wa.z, xa.w * rs * wa.w);
    dst[2] = pkbf(xb.x * rs * wb.x, xb.y * rs * wb.y);
    dst[3] = pkbf(xb.z * rs * wb.z, xb.w * rs * wb.w);
  }

  // router logits in fp32: 16 threads per expert, 16 experts
  __shared__ float slog[cE];
  if (tid < cE) slog[tid] = 0.f;
  __syncthreads();
  int e = tid & 15;
  int part = tid >> 4;
  const float* gwe = gw + (size_t)e * cH;
  float p = 0.f;
  for (int i = 0; i < cH / 16; i++) {
    int h = part + i * 16;
    p += (xr[h] * rs * nw[h]) * gwe[h];
  }
  atomicAdd(&slog[e], p);
  __syncthreads();

  if (tid == 0) {
    float mx = -1e30f;
    for (int i = 0; i < cE; i++) mx = fmaxf(mx, slog[i]);
    float pr[cE];
    float sum = 0.f;
    for (int i = 0; i < cE; i++) { pr[i] = __expf(slog[i] - mx); sum += pr[i]; }
    int i0 = 0; float b0 = -1.f;
    for (int i = 0; i < cE; i++) if (pr[i] > b0) { b0 = pr[i]; i0 = i; }
    int i1 = 0; float b1 = -1.f;
    for (int i = 0; i < cE; i++) { if (i == i0) continue; if (pr[i] > b1) { b1 = pr[i]; i1 = i; } }
    float wsum = b0 + b1;               // softmax denom cancels in renorm
    float w0 = b0 / wsum, w1 = b1 / wsum;
    topkw[t * 2] = w0;       topkw[t * 2 + 1] = w1;
    topkids[t * 2] = i0;     topkids[t * 2 + 1] = i1;
    out_topkw[t * 2] = w0;   out_topkw[t * 2 + 1] = w1;
    out_topkids[t * 2] = i0; out_topkids[t * 2 + 1] = i1;
    atomicAdd(&counts[i0], 1);
    atomicAdd(&counts[i1], 1);
  }
}

// ------------------------------------------- expert tile scan + row scatter
__global__ void scan_kernel(const int* __restrict__ counts, int* offsets,
                            int* cursor, int* tile_e, int* tile_m, int* tile_r,
                            int* num_tiles) {
  if (threadIdx.x == 0) {
    int off = 0, nt = 0;
    for (int e = 0; e < cE; e++) {
      offsets[e] = off;
      cursor[e] = off;
      int c = counts[e];
      for (int s = 0; s < c; s += BM) {
        tile_e[nt] = e;
        tile_m[nt] = off + s;
        int rem = c - s;
        tile_r[nt] = rem < BM ? rem : BM;
        nt++;
      }
      off += c;
    }
    num_tiles[0] = nt;
  }
}

__global__ void scatter_kernel(const int* __restrict__ ids, int* cursor,
                               int* __restrict__ rowlist) {
  int t = blockIdx.x * blockDim.x + threadIdx.x;
  if (t < cT) {
#pragma unroll
    for (int k = 0; k < cK; k++) {
      int e = ids[t * cK + k];
      int pos = atomicAdd(&cursor[e], 1);
      rowlist[pos] = t * cK + k;      // slot encodes (token, k)
    }
  }
}

// ------------------------------------------------------------ WMMA GEMM core
// C[m,n] = sum_k A[m,k] * B[k,n], A bf16, B fp32 (converted inline), fp32 acc.
// Double-buffered LDS: global loads for tile i+1 overlap WMMA on tile i.
// DUAL: two B matrices (gate/up) sharing one A fetch, silu-fused epilogue.
// TRANSB: B given as [N,K] row-major (i.e. W.T matmul).
// EPI: 1 = silu(g)*u -> bf16 out ; 2 = top-k weight * atomicAdd into moe ;
//      3 = residual + moe + acc -> f32 out.
template <bool DUAL, bool TRANSB, int EPI, bool GATHER_A>
__global__ void gemm_wmma_kernel(
    const unsigned short* __restrict__ A, int lda,
    const float* __restrict__ Bg, const float* __restrict__ Bu,
    size_t expert_stride, int ldb, int Kdim,
    const int* __restrict__ rowlist,
    const int* __restrict__ tile_e, const int* __restrict__ tile_m,
    const int* __restrict__ tile_r, const int* __restrict__ num_tiles,
    void* __restrict__ outp, int ldo,
    const float* __restrict__ topkw, float* __restrict__ moe_acc,
    const float* __restrict__ residual, const float* __restrict__ moe_in) {
  constexpr int NB = DUAL ? 2 : 1;

  int tileY = blockIdx.y;
  int e = 0, mstart, rows;
  if (tile_e) {                      // grouped (per-expert) mode
    if (tileY >= num_tiles[0]) return;
    e = tile_e[tileY];
    mstart = tile_m[tileY];
    rows = tile_r[tileY];
  } else {                           // dense mode
    mstart = tileY * BM;
    rows = BM;
  }
  const float* Bgp = Bg + (size_t)e * expert_stride;
  const float* Bup = DUAL ? (Bu + (size_t)e * expert_stride) : nullptr;
  int n0 = blockIdx.x * BN;

  __shared__ unsigned short As[2][BM * LDSP];
  __shared__ unsigned short Bs[2][NB][BN * LDSP];

  int tid = threadIdx.x;
  int lane = tid & 31, wave = tid >> 5;
  int wm0 = (wave >> 1) * 16;        // 4 waves along M
  int wn0 = (wave & 1) * 32;         // 2 waves along N, 2 sub-tiles each
  int l15 = lane & 15, lhi = lane >> 4;

  v8f acc0 = {}, acc1 = {}, uacc0 = {}, uacc1 = {};

  // A staging assignment: 4 threads per row, 8 halves each
  int a_m = tid >> 2;
  int a_kseg = (tid & 3) * 8;
  bool a_valid = a_m < rows;
  size_t a_row = 0;
  if (a_valid) {
    int gm = mstart + a_m;
    a_row = GATHER_A ? (size_t)(rowlist[gm] >> 1) : (size_t)gm;
  }

  auto stage = [&](int buf, int k0) {
    // ---- A tile (bf16 global -> LDS)
    {
      uint4 v = make_uint4(0u, 0u, 0u, 0u);
      if (a_valid) v = *(const uint4*)(A + a_row * lda + (k0 + a_kseg));
      unsigned int* d = (unsigned int*)&As[buf][a_m * LDSP + a_kseg];
      d[0] = v.x; d[1] = v.y; d[2] = v.z; d[3] = v.w;
    }
    // ---- B tile(s) (fp32 global -> bf16 LDS, stored [n][k])
    if (!TRANSB) {
      int kk = tid & 31;
      int nseg = (tid >> 5) * 8;
      const float* s0 = Bgp + (size_t)(k0 + kk) * ldb + (n0 + nseg);
      if (k0 + BK < Kdim) __builtin_prefetch(s0 + (size_t)BK * ldb, 0, 1);
      float4 f0 = *(const float4*)(s0);
      float4 f1 = *(const float4*)(s0 + 4);
      unsigned int p0 = pkbf(f0.x, f0.y), p1 = pkbf(f0.z, f0.w);
      unsigned int p2 = pkbf(f1.x, f1.y), p3 = pkbf(f1.z, f1.w);
      unsigned short* bs = &Bs[buf][0][nseg * LDSP + kk];
      bs[0 * LDSP] = (unsigned short)p0; bs[1 * LDSP] = (unsigned short)(p0 >> 16);
      bs[2 * LDSP] = (unsigned short)p1; bs[3 * LDSP] = (unsigned short)(p1 >> 16);
      bs[4 * LDSP] = (unsigned short)p2; bs[5 * LDSP] = (unsigned short)(p2 >> 16);
      bs[6 * LDSP] = (unsigned short)p3; bs[7 * LDSP] = (unsigned short)(p3 >> 16);
      if (DUAL) {
        const float* s1 = Bup + (size_t)(k0 + kk) * ldb + (n0 + nseg);
        float4 g0 = *(const float4*)(s1);
        float4 g1 = *(const float4*)(s1 + 4);
        unsigned int q0 = pkbf(g0.x, g0.y), q1 = pkbf(g0.z, g0.w);
        unsigned int q2 = pkbf(g1.x, g1.y), q3 = pkbf(g1.z, g1.w);
        unsigned short* bu = &Bs[buf][NB - 1][nseg * LDSP + kk];
        bu[0 * LDSP] = (unsigned short)q0; bu[1 * LDSP] = (unsigned short)(q0 >> 16);
        bu[2 * LDSP] = (unsigned short)q1; bu[3 * LDSP] = (unsigned short)(q1 >> 16);
        bu[4 * LDSP] = (unsigned short)q2; bu[5 * LDSP] = (unsigned short)(q2 >> 16);
        bu[6 * LDSP] = (unsigned short)q3; bu[7 * LDSP] = (unsigned short)(q3 >> 16);
      }
    } else {
      int nl = tid & 63;
      int ks = (tid >> 6) * 8;
      const float* s0 = Bgp + (size_t)(n0 + nl) * ldb + (k0 + ks);
      if (k0 + BK < Kdim) __builtin_prefetch(s0 + BK, 0, 1);
      float4 f0 = *(const float4*)(s0);
      float4 f1 = *(const float4*)(s0 + 4);
      unsigned int* d = (unsigned int*)&Bs[buf][0][nl * LDSP + ks];
      d[0] = pkbf(f0.x, f0.y);
      d[1] = pkbf(f0.z, f0.w);
      d[2] = pkbf(f1.x, f1.y);
      d[3] = pkbf(f1.z, f1.w);
      if (DUAL) {
        const float* s1 = Bup + (size_t)(n0 + nl) * ldb + (k0 + ks);
        float4 g0 = *(const float4*)(s1);
        float4 g1 = *(const float4*)(s1 + 4);
        unsigned int* du = (unsigned int*)&Bs[buf][NB - 1][nl * LDSP + ks];
        du[0] = pkbf(g0.x, g0.y);
        du[1] = pkbf(g0.z, g0.w);
        du[2] = pkbf(g1.x, g1.y);
        du[3] = pkbf(g1.z, g1.w);
      }
    }
  };

  stage(0, 0);
  int cur = 0;
  for (int k0 = 0; k0 < Kdim; k0 += BK) {
    __syncthreads();                       // staging of `cur` complete
    if (k0 + BK < Kdim) stage(cur ^ 1, k0 + BK);  // overlap next-tile loads

    // ---- gather fragments (contiguous k = 16*lhi + s, same for A and B) and accumulate
    FragU af;
#pragma unroll
    for (int p = 0; p < 8; p++) {
      int k = lhi * 16 + 2 * p;                          // A 16x32, contiguous k order
      af.u[p] = *(const unsigned int*)&As[cur][(wm0 + l15) * LDSP + k];
    }
#pragma unroll
    for (int nt = 0; nt < 2; nt++) {
      int nl = wn0 + nt * 16 + l15;
      FragU bg;
#pragma unroll
      for (int p = 0; p < 8; p++)                      // B 32x16, same contiguous k order as A
        bg.u[p] = *(const unsigned int*)&Bs[cur][0][nl * LDSP + lhi * 16 + 2 * p];
      v8f& acc = (nt == 0) ? acc0 : acc1;
      acc = __builtin_amdgcn_wmma_f32_16x16x32_bf16(
          false, af.v, false, bg.v, (short)0, acc, false, false);
      if (DUAL) {
        FragU bu;
#pragma unroll
        for (int p = 0; p < 8; p++)
          bu.u[p] = *(const unsigned int*)&Bs[cur][NB - 1][nl * LDSP + lhi * 16 + 2 * p];
        v8f& ua = (nt == 0) ? uacc0 : uacc1;
        ua = __builtin_amdgcn_wmma_f32_16x16x32_bf16(
            false, af.v, false, bu.v, (short)0, ua, false, false);
      }
    }
    cur ^= 1;
  }

  // ---- epilogue (C/D layout: VGPR r -> m = wm0 + r + lhi*8, n = l15)
#pragma unroll
  for (int nt = 0; nt < 2; nt++) {
    int n = n0 + wn0 + nt * 16 + l15;
    v8f g = (nt == 0) ? acc0 : acc1;
    v8f u = (nt == 0) ? uacc0 : uacc1;
#pragma unroll
    for (int r = 0; r < 8; r++) {
      int ml = wm0 + r + lhi * 8;
      if (ml >= rows) continue;
      int gm = mstart + ml;
      float gv = g[r];
      if (EPI == 1) {
        float sv = gv / (1.f + __expf(-gv)) * u[r];   // silu(g) * u
        ((unsigned short*)outp)[(size_t)gm * ldo + n] = f2bf(sv);
      } else if (EPI == 2) {
        int slot = rowlist[gm];
        int tkn = slot >> 1;
        atomicAdd(&moe_acc[(size_t)tkn * ldo + n], gv * topkw[slot]);
      } else if (EPI == 3) {
        size_t idx = (size_t)gm * ldo + n;
        ((float*)outp)[idx] = residual[idx] + moe_in[idx] + gv;
      }
    }
  }
}

// ---------------------------------------------------------------- launcher
extern "C" void kernel_launch(void* const* d_in, const int* in_sizes, int n_in,
                              void* d_out, int out_size, void* d_ws,
                              size_t ws_size, hipStream_t stream) {
  (void)in_sizes; (void)n_in; (void)out_size; (void)ws_size;
  const float* x   = (const float*)d_in[0];   // [T,H]
  const float* nw  = (const float*)d_in[1];   // [H]
  const float* gw  = (const float*)d_in[2];   // [E,H]
  const float* w13 = (const float*)d_in[3];   // [E,H,2I]
  const float* w2  = (const float*)d_in[4];   // [E,I,H]
  const float* shg = (const float*)d_in[5];   // [I,H]
  const float* shu = (const float*)d_in[6];   // [I,H]
  const float* shd = (const float*)d_in[7];   // [H,I]

  char* ws = (char*)d_ws;
  size_t off = 0;
  auto take = [&](size_t bytes) {
    char* p = ws + off;
    off = (off + bytes + 255) & ~(size_t)255;
    return p;
  };
  unsigned short* xn    = (unsigned short*)take((size_t)cT * cH * 2);
  float* topkw          = (float*)take((size_t)cT * cK * 4);
  int* topkids          = (int*)take((size_t)cT * cK * 4);
  int* counts           = (int*)take(cE * 4);
  int* offsets          = (int*)take(cE * 4);
  int* cursor           = (int*)take(cE * 4);
  int* numtiles         = (int*)take(4);
  int* tile_e           = (int*)take(MAXTILES * 4);
  int* tile_m           = (int*)take(MAXTILES * 4);
  int* tile_r           = (int*)take(MAXTILES * 4);
  int* rowlist          = (int*)take((size_t)cT * cK * 4);
  unsigned short* act   = (unsigned short*)take((size_t)cT * cK * cI * 2);
  float* moe            = (float*)take((size_t)cT * cH * 4);
  unsigned short* moebf = (unsigned short*)take((size_t)cT * cH * 2);
  unsigned short* sg    = (unsigned short*)take((size_t)cT * cI * 2);

  float* out_topkw = (float*)d_out + (size_t)cT * cH;
  int* out_topkids = (int*)d_out + (size_t)cT * cH + (size_t)cT * cK;

  init_kernel<<<1, 64, 0, stream>>>(counts);
  zero_kernel<<<(cT * cH / 4) / 256, 256, 0, stream>>>(moe, (size_t)cT * cH / 4);
  norm_gate_kernel<<<cT, 256, 0, stream>>>(x, nw, gw, xn, topkw, topkids,
                                           out_topkw, out_topkids, counts);
  scan_kernel<<<1, 32, 0, stream>>>(counts, offsets, cursor, tile_e, tile_m,
                                    tile_r, numtiles);
  scatter_kernel<<<cT / 256, 256, 0, stream>>>(topkids, cursor, rowlist);

  // grouped GEMM1: act = silu(x @ Wg) * (x @ Wu), per expert (w13 = [H,2I])
  gemm_wmma_kernel<true, false, 1, true>
      <<<dim3(cI / BN, MAXTILES), 256, 0, stream>>>(
          xn, cH, w13, w13 + cI, (size_t)cH * 2 * cI, 2 * cI, cH,
          rowlist, tile_e, tile_m, tile_r, numtiles,
          act, cI, nullptr, nullptr, nullptr, nullptr);

  // grouped GEMM2: moe += topk_w * (act @ w2[e]), atomic combine
  gemm_wmma_kernel<false, false, 2, false>
      <<<dim3(cH / BN, MAXTILES), 256, 0, stream>>>(
          act, cI, w2, nullptr, (size_t)cI * cH, cH, cI,
          rowlist, tile_e, tile_m, tile_r, numtiles,
          nullptr, cH, topkw, moe, nullptr, nullptr);

  f32_to_bf16_kernel<<<(cT * cH / 2) / 256, 256, 0, stream>>>(
      moe, (unsigned int*)moebf, (size_t)cT * cH / 2);

  // dense GEMM3: sg = silu(moe @ shg.T) * (moe @ shu.T)   (B transposed)
  gemm_wmma_kernel<true, true, 1, false>
      <<<dim3(cI / BN, cT / BM), 256, 0, stream>>>(
          moebf, cH, shg, shu, 0, cH, cH,
          nullptr, nullptr, nullptr, nullptr, nullptr,
          sg, cI, nullptr, nullptr, nullptr, nullptr);

  // dense GEMM4: out = residual + moe + sg @ shd.T        (B transposed)
  gemm_wmma_kernel<false, true, 3, false>
      <<<dim3(cH / BN, cT / BM), 256, 0, stream>>>(
          sg, cI, shd, nullptr, 0, cI, cI,
          nullptr, nullptr, nullptr, nullptr, nullptr,
          d_out, cH, nullptr, nullptr, x, moe);
}
